// row_attention_block_63136019251725
// MI455X (gfx1250) — matrix operand order repaired and checked by reading, not compiled, not run on hardware
//
#include <hip/hip_runtime.h>

// MI455X / gfx1250 row-attention block using v_wmma_f32_16x16x32_f16.
// One workgroup (256 threads = 8 wave32) per (b,d,h) site; grid = B*D*H = 4096.
// Shapes fixed by reference: B=2,C=64,D=32,H=64,W=96,qk=32.

typedef _Float16 v16h __attribute__((ext_vector_type(16)));
typedef _Float16 v8h  __attribute__((ext_vector_type(8)));
typedef float    v8f  __attribute__((ext_vector_type(8)));

#define C_DIM   64
#define W_DIM   96
#define QK_DIM  32
#define C_STRIDE 196608      // D*H*W = 32*64*96
#define B_STRIDE 12582912    // C*D*H*W
#define D_STRIDE 6144        // H*W

// LDS strides in halves (all multiples of 8 halves = 16B for aligned b128 frags)
#define XT_S 72    // Xt[w][c]   96 x 72
#define WM_S 72    // W[q][c]    rows x 72
#define QT_S 40    // Qt/Kt[i][q] 96 x 40
#define VH_S 104   // Vh[c][j]   64 x 104
#define P_S  104   // P[i][j]    96 x 104  (aliases Xt+weights region)

__device__ __forceinline__ v8f vzero8f() {
  v8f z = {0.f, 0.f, 0.f, 0.f, 0.f, 0.f, 0.f, 0.f};
  return z;
}

// A-fragment (16x32 f16): lane holds row m; K halves {kb..kb+7, 16+kb..16+kb+7}
__device__ __forceinline__ v16h frag_a(const _Float16* row, int kb) {
  v8h lo = *(const v8h*)(row + kb);
  v8h hi = *(const v8h*)(row + 16 + kb);
  return __builtin_shufflevector(lo, hi, 0, 1, 2, 3, 4, 5, 6, 7,
                                 8, 9, 10, 11, 12, 13, 14, 15);
}

// B-fragment (32x16 f16): lane holds col n; K halves {h16/2..h16/2+7, 16+h16/2..16+h16/2+7}
__device__ __forceinline__ v16h frag_b(const _Float16* row, int h16) {
  v8h lo = *(const v8h*)(row + (h16 >> 1));
  v8h hi = *(const v8h*)(row + 16 + (h16 >> 1));
  return __builtin_shufflevector(lo, hi, 0, 1, 2, 3, 4, 5, 6, 7,
                                 8, 9, 10, 11, 12, 13, 14, 15);
}

// sinusoidal PE: pe[row][p]; even row -> sin(p * 10000^{-row/d}), odd -> cos with row-1
__device__ __forceinline__ float pe_val(int row, int p, float inv_d) {
  float i2 = (float)(row & ~1);
  float dv = __expf(-i2 * 9.210340371976184f * inv_d);  // ln(10000)
  float ang = (float)p * dv;
  return (row & 1) ? __cosf(ang) : __sinf(ang);
}

__global__ __launch_bounds__(256) void row_attn_wmma_kernel(
    const float* __restrict__ x,
    const float* __restrict__ Wq, const float* __restrict__ bq,
    const float* __restrict__ Wk, const float* __restrict__ bk,
    const float* __restrict__ Wv, const float* __restrict__ bv,
    const float* __restrict__ gamma,
    float* __restrict__ out) {
  // Static LDS: 30464 halves = 60928 bytes.
  //  [0,6912)      Xt   96x72      (phase A/B)   -- aliased by P[0,9984) from phase C on
  //  [6912,9216)   Wqh  32x72
  //  [9216,11520)  Wkh  32x72
  //  [11520,16128) Wvh  64x72
  //  [16128,19968) Qt   96x40  (Q transposed: [i][q])
  //  [19968,23808) Kt   96x40
  //  [23808,30464) Vh   64x104
  __shared__ __align__(16) _Float16 smem[30464];
  _Float16* Xt  = smem;
  _Float16* Wqh = smem + 6912;
  _Float16* Wkh = smem + 9216;
  _Float16* Wvh = smem + 11520;
  _Float16* P   = smem;  // alias: live only after Xt/W* are dead
  _Float16* Qt  = smem + 16128;
  _Float16* Kt  = smem + 19968;
  _Float16* Vh  = smem + 23808;

  const int tid  = threadIdx.x;
  const int wv   = tid >> 5;     // wave id 0..7
  const int lane = tid & 31;
  const int hi   = lane >> 4;    // half-wave select
  const int mn   = lane & 15;    // m (A rows) / n (B,D cols)
  const int kb   = hi * 8;
  const int h16  = hi * 16;

  const int bid = blockIdx.x;            // B*D*H = 4096
  const int hh  = bid & 63;
  const int dd  = (bid >> 6) & 31;
  const int bb  = bid >> 11;
  const int xbase = bb * B_STRIDE + dd * D_STRIDE + hh * W_DIM;

  // ---------------- Phase A: stage X (transposed, f16) and weights -------------
  for (int e = tid; e < C_DIM * W_DIM; e += 256) {
    int c = e / W_DIM;
    int w = e - c * W_DIM;
    Xt[w * XT_S + c] = (_Float16)x[xbase + c * C_STRIDE + w];
  }
  for (int e = tid; e < QK_DIM * C_DIM; e += 256) {
    int r = e >> 6, c = e & 63;
    Wqh[r * WM_S + c] = (_Float16)Wq[e];
    Wkh[r * WM_S + c] = (_Float16)Wk[e];
  }
  for (int e = tid; e < C_DIM * C_DIM; e += 256) {
    int r = e >> 6, c = e & 63;
    Wvh[r * WM_S + c] = (_Float16)Wv[e];
  }
  __syncthreads();

  // ---------------- Phase B: projections Q,K,V (+bias +PE), 48 tiles -----------
  // tiles 0..11: Q (2x6), 12..23: K, 24..47: V (4x6). Wave-uniform distribution.
  for (int id = wv; id < 48; id += 8) {
    if (id < 24) {
      const bool isQ = (id < 12);
      const int t  = isQ ? id : id - 12;
      const int rb = (t / 6) * 16;           // q-row block
      const int it = t - (t / 6) * 6;        // i (position) block
      const _Float16* Wm = isQ ? Wqh : Wkh;
      const float* bias  = isQ ? bq : bk;
      _Float16* dst      = isQ ? Qt : Kt;
      v8f acc = vzero8f();
#pragma unroll
      for (int kc = 0; kc < 2; ++kc) {       // K over C=64
        v16h a  = frag_a(Wm + (rb + mn) * WM_S + kc * 32, kb);
        v16h bm = frag_b(Xt + (it * 16 + mn) * XT_S + kc * 32, h16);
        acc = __builtin_amdgcn_wmma_f32_16x16x32_f16(false, a, false, bm,
                                                     (short)0, acc, false, false);
      }
      v8h hs;
#pragma unroll
      for (int r = 0; r < 8; ++r) {
        int row = rb + hi * 8 + r;           // q index
        float v = acc[r] + bias[row] + pe_val(row, it * 16 + mn, 1.0f / 32.0f);
        hs[r] = (_Float16)v;
      }
      // store transposed: Qt[i][q], 8 contiguous halves per lane (ds_store_b128)
      *(v8h*)(dst + (it * 16 + mn) * QT_S + rb + hi * 8) = hs;
    } else {
      const int t  = id - 24;
      const int rb = (t / 6) * 16;           // c-row block
      const int it = t - (t / 6) * 6;
      v8f acc = vzero8f();
#pragma unroll
      for (int kc = 0; kc < 2; ++kc) {
        v16h a  = frag_a(Wvh + (rb + mn) * WM_S + kc * 32, kb);
        v16h bm = frag_b(Xt + (it * 16 + mn) * XT_S + kc * 32, h16);
        acc = __builtin_amdgcn_wmma_f32_16x16x32_f16(false, a, false, bm,
                                                     (short)0, acc, false, false);
      }
#pragma unroll
      for (int r = 0; r < 8; ++r) {
        int row = rb + hi * 8 + r;           // channel c
        float v = acc[r] + bv[row] + pe_val(row, it * 16 + mn, 1.0f / 64.0f);
        Vh[row * VH_S + it * 16 + mn] = (_Float16)v;
      }
    }
  }
  __syncthreads();

  // ---------------- Phase C: S = Q^T K  (one WMMA per 16x16 tile, K=qk=32) -----
  for (int id = wv; id < 36; id += 8) {
    const int it = id / 6;
    const int jt = id - it * 6;
    v16h a  = frag_a(Qt + (it * 16 + mn) * QT_S, kb);   // A(m=i, k=q)
    v16h bm = frag_b(Kt + (jt * 16 + mn) * QT_S, h16);  // B(k=q, n=j)
    v8f acc = vzero8f();
    acc = __builtin_amdgcn_wmma_f32_16x16x32_f16(false, a, false, bm,
                                                 (short)0, acc, false, false);
#pragma unroll
    for (int r = 0; r < 8; ++r)
      P[(it * 16 + hi * 8 + r) * P_S + jt * 16 + mn] = (_Float16)acc[r];
  }
  __syncthreads();

  // ---------------- Phase D: row softmax in place on P -------------------------
  for (int row = tid; row < W_DIM; row += 256) {
    _Float16* pr = P + row * P_S;
    float mx = -3.0e38f;
    for (int j = 0; j < W_DIM; ++j) mx = fmaxf(mx, (float)pr[j]);
    float sum = 0.f;
    for (int j = 0; j < W_DIM; ++j) {
      float e = __expf((float)pr[j] - mx);
      sum += e;
      pr[j] = (_Float16)e;
    }
    float inv = 1.0f / sum;
    for (int j = 0; j < W_DIM; ++j) pr[j] = (_Float16)((float)pr[j] * inv);
  }
  __syncthreads();

  // ---------------- Phase E: O = V * P^T, fused residual write -----------------
  const float g = gamma[0];
  for (int id = wv; id < 24; id += 8) {
    const int ct = id / 6;
    const int it = id - ct * 6;
    v8f acc = vzero8f();
#pragma unroll
    for (int ks = 0; ks < 3; ++ks) {         // K over j = 96
      v16h a  = frag_a(Vh + (ct * 16 + mn) * VH_S + ks * 32, kb);  // A(m=c,k=j)
      v16h bm = frag_b(P  + (it * 16 + mn) * P_S  + ks * 32, h16); // B(k=j,n=i)
      acc = __builtin_amdgcn_wmma_f32_16x16x32_f16(false, a, false, bm,
                                                   (short)0, acc, false, false);
    }
    const int i = it * 16 + mn;
#pragma unroll
    for (int r = 0; r < 8; ++r) {
      int c  = ct * 16 + hi * 8 + r;
      int gi = xbase + c * C_STRIDE + i;
      out[gi] = g * acc[r] + x[gi];
    }
  }
}

extern "C" void kernel_launch(void* const* d_in, const int* in_sizes, int n_in,
                              void* d_out, int out_size, void* d_ws, size_t ws_size,
                              hipStream_t stream) {
  const float* x  = (const float*)d_in[0];
  const float* Wq = (const float*)d_in[1];
  const float* bq = (const float*)d_in[2];
  const float* Wk = (const float*)d_in[3];
  const float* bk = (const float*)d_in[4];
  const float* Wv = (const float*)d_in[5];
  const float* bv = (const float*)d_in[6];
  const float* gm = (const float*)d_in[7];
  float* out = (float*)d_out;
  (void)n_in; (void)out_size; (void)d_ws; (void)ws_size;

  const int grid = in_sizes[0] / (C_DIM * W_DIM);  // B*D*H = 4096
  row_attn_wmma_kernel<<<grid, 256, 0, stream>>>(x, Wq, bq, Wk, bk, Wv, bv, gm, out);
}
